// MoDInfiniTransformer_7645041787085
// MI455X (gfx1250) — matrix operand order repaired and checked by reading, not compiled, not run on hardware
//
#include <hip/hip_runtime.h>
#include <cstdint>

// ---------------- problem constants ----------------
#define BB     4
#define SS     16384
#define DD     512
#define DHID   2048
#define NH     8
#define DKV    64
#define SEGF   2048
#define SEGC   512          // MoD capacity per segment == attention segment len
#define NSEGF  8            // SS / SEG_FULL
#define SSEL   4096         // selected tokens per batch
#define MTOT   (BB * SSEL)  // 16384 rows through the transformer

typedef __bf16 bf16;
typedef __attribute__((ext_vector_type(16))) __bf16 v16bf;
typedef __attribute__((ext_vector_type(8)))  float  v8f;
typedef __attribute__((ext_vector_type(4)))  unsigned int u32x4;
typedef __attribute__((ext_vector_type(8)))  int i32x8;
typedef __attribute__((ext_vector_type(4)))  int i32x4;

#if __has_builtin(__builtin_amdgcn_tensor_load_to_lds)
#define USE_TDM 1
#else
#define USE_TDM 0
#endif

// ---------------- WMMA helpers (CDNA5, wave32) ----------------
__device__ __forceinline__ v8f wmma_bf16f32(v16bf a, v16bf b, v8f c) {
  return __builtin_amdgcn_wmma_f32_16x16x32_bf16(false, a, false, b, (short)0, c,
                                                 false, false);
}

__device__ __forceinline__ float elu1f(float x) {      // elu(x)+1
  return x > 0.f ? x + 1.f : __expf(x);
}
__device__ __forceinline__ float gelu_exact(float x) {
  return 0.5f * x * (1.f + erff(x * 0.70710678118654752f));
}

// A-matrix 16x32 bf16 fragment from row-major src (ld elems/row).
// Lane L: row = L%16; lane-half selects K {0-7,16-23} vs {8-15,24-31}.
// Both K-groups are contiguous -> two b128 loads, no scalar packing.
struct a_frag_bits { uint4 lo, hi; };
__device__ __forceinline__ v16bf load_a16x32(const bf16* src, int ld) {
  int lane = threadIdx.x & 31;
  const bf16* rp = src + (size_t)(lane & 15) * ld + ((lane >> 4) << 3);
  a_frag_bits p;
  p.lo = *(const uint4*)rp;         // K 0..7   (+half*8)
  p.hi = *(const uint4*)(rp + 16);  // K 16..23 (+half*8)
  return __builtin_bit_cast(v16bf, p);
}

// B-matrix 32x16 fragment from transposed [n][k] storage: for lane (n=L%16,
// half = L/16) K {0-7,16-23} (+half*8), matching A -> two b128 loads.
__device__ __forceinline__ v16bf load_b32x16_tr(const bf16* src, int ldk) {
  int lane = threadIdx.x & 31;
  const bf16* cp = src + (size_t)(lane & 15) * ldk + ((lane >> 4) << 3);
  a_frag_bits p;
  p.lo = *(const uint4*)cp;
  p.hi = *(const uint4*)(cp + 16);
  return __builtin_bit_cast(v16bf, p);
}

// ---------------- Tensor Data Mover: 2D tile -> LDS ----------------
#if USE_TDM
// D# per CDNA5 ISA ch8: group0 = {flags, lds_addr, global_addr, type=2},
// group1 = {wg_mask/data_size, tensor dims, tile dims, dim0 stride}.
// 6-arg builtin form: (g0, g1, g2, g3, g4, cpol); groups 2-4 unused for 2D.
__device__ __forceinline__ void tdm_load_2d(unsigned lds_off, const void* gaddr,
                                            unsigned tile_d0, unsigned tile_d1,
                                            unsigned tensor_d0, unsigned tensor_d1,
                                            unsigned stride0) {
  unsigned long long ga = (unsigned long long)gaddr;
  u32x4 g0;
  g0[0] = 1u;                                             // count=1, user mode
  g0[1] = lds_off;                                        // LDS byte address
  g0[2] = (unsigned)ga;                                   // global addr [31:0]
  g0[3] = (unsigned)((ga >> 32) & 0x1FFFFFFu) | (2u << 30);  // [56:32] | type=2
  i32x8 g1;
  g1[0] = (int)(1u << 16);                                // data_size = 2 bytes
  g1[1] = (int)((tensor_d0 & 0xFFFFu) << 16);             // tensor_dim0[15:0]
  g1[2] = (int)((tensor_d0 >> 16) | ((tensor_d1 & 0xFFFFu) << 16));
  g1[3] = (int)((tensor_d1 >> 16) | (tile_d0 << 16));     // tile_dim0
  g1[4] = (int)tile_d1;                                   // tile_dim1, dim2=0
  g1[5] = (int)stride0;                                   // dim0 stride [31:0]
  g1[6] = 0;
  g1[7] = 0;
  i32x4 z4 = {0, 0, 0, 0};
  i32x8 z8 = {0, 0, 0, 0, 0, 0, 0, 0};
  __builtin_amdgcn_tensor_load_to_lds(g0, g1, z4, z4, z8, 0);
}
#endif

// Manual fallbacks (and non-TDM toolchains): b128 cooperative copies.
__device__ __forceinline__ void load_rm_chunk(const bf16* g, bf16* lds,
                                              size_t row0, size_t hoff) {
  int tid = threadIdx.x;
  int row = tid >> 1, cg = (tid & 1) * 32;             // 128 rows x 64 cols
  const uint4* s = (const uint4*)(g + (row0 + row) * DD + hoff + cg);
  uint4* d = (uint4*)(lds + row * DKV + cg);
#pragma unroll
  for (int i = 0; i < 4; ++i) d[i] = s[i];
}
__device__ __forceinline__ void load_tr_chunk(const bf16* g, bf16* lds,
                                              size_t tok0, size_t hoff) {
  int tid = threadIdx.x;
  int row = tid >> 2, c0 = (tid & 3) * 32;             // 64 rows x 128 cols
  const uint4* s = (const uint4*)(g + (hoff + row) * (size_t)MTOT + tok0 + c0);
  uint4* d = (uint4*)(lds + row * 128 + c0);
#pragma unroll
  for (int i = 0; i < 4; ++i) d[i] = s[i];
}

// ---------------- small utility kernels ----------------
// f32 -> bf16 with transpose: out[n*K + k] = in[k*N + n]
__global__ void convert_transpose_kernel(const float* __restrict__ in,
                                         unsigned short* __restrict__ out,
                                         int K, int N) {
  int idx = blockIdx.x * 256 + threadIdx.x;
  if (idx < K * N) {
    int k = idx / N, n = idx % N;
    ((bf16*)out)[(size_t)n * K + k] = (bf16)in[idx];
  }
}

// Router: scores = x @ w_samp + b_samp.  One wave per token, b128 loads.
__global__ __launch_bounds__(256) void router_kernel(
    const float* __restrict__ x, const float* __restrict__ w,
    const float* __restrict__ bsc, float* __restrict__ scores) {
  int wv = threadIdx.x >> 5, lane = threadIdx.x & 31;
  size_t tok = (size_t)blockIdx.x * 8 + wv;
  const float* xp = x + tok * DD;
  float s = 0.f;
#pragma unroll
  for (int j = 0; j < 4; ++j) {
    float4 a = *(const float4*)(xp + lane * 4 + j * 128);
    float4 ww = *(const float4*)(w + lane * 4 + j * 128);
    s += a.x * ww.x + a.y * ww.y + a.z * ww.z + a.w * ww.w;
  }
#pragma unroll
  for (int m = 16; m >= 1; m >>= 1) s += __shfl_xor(s, m, 32);
  if (lane == 0) scores[tok] = s + bsc[0];
}

// Per (batch, full-segment): top-512 of 2048 by score, indices re-sorted ascending.
__global__ __launch_bounds__(1024) void topk_kernel(const float* __restrict__ scores,
                                                    int* __restrict__ flat) {
  __shared__ float sc[SEGF];
  __shared__ int idx[SEGF];
  __shared__ int sel[SEGC];
  int b = blockIdx.x >> 3, seg = blockIdx.x & 7;
  const float* sp = scores + (size_t)b * SS + (size_t)seg * SEGF;
  for (int i = threadIdx.x; i < SEGF; i += 1024) { sc[i] = sp[i]; idx[i] = i; }
  __syncthreads();
  for (int k = 2; k <= SEGF; k <<= 1) {
    for (int j = k >> 1; j > 0; j >>= 1) {
      for (int t = threadIdx.x; t < SEGF; t += 1024) {
        int ixj = t ^ j;
        if (ixj > t) {
          bool desc = ((t & k) == 0);
          float a = sc[t], c = sc[ixj];
          if (desc ? (a < c) : (a > c)) {
            sc[t] = c; sc[ixj] = a;
            int ti = idx[t]; idx[t] = idx[ixj]; idx[ixj] = ti;
          }
        }
      }
      __syncthreads();
    }
  }
  for (int i = threadIdx.x; i < SEGC; i += 1024) sel[i] = idx[i];
  __syncthreads();
  for (int k = 2; k <= SEGC; k <<= 1) {
    for (int j = k >> 1; j > 0; j >>= 1) {
      int t = threadIdx.x;
      if (t < SEGC) {
        int ixj = t ^ j;
        if (ixj > t) {
          bool up = ((t & k) == 0);
          int a = sel[t], c = sel[ixj];
          if (up ? (a > c) : (a < c)) { sel[t] = c; sel[ixj] = a; }
        }
      }
      __syncthreads();
    }
  }
  for (int i = threadIdx.x; i < SEGC; i += 1024)
    flat[(size_t)b * SSEL + seg * SEGC + i] = seg * SEGF + sel[i];
}

__global__ __launch_bounds__(128) void gather_kernel(
    const float* __restrict__ x, const int* __restrict__ flat,
    unsigned short* __restrict__ xsel) {
  int r = blockIdx.x;
  int b = r / SSEL;
  int src = flat[r];
  const float4* xp = (const float4*)(x + ((size_t)b * SS + src) * DD);
  bf16* op = (bf16*)xsel + (size_t)r * DD;
  float4 vq = xp[threadIdx.x];
  bf16 o4[4] = {(bf16)vq.x, (bf16)vq.y, (bf16)vq.z, (bf16)vq.w};
  *(uint2*)(op + threadIdx.x * 4) = *(const uint2*)o4;
}

// ---------------- bf16 WMMA GEMM: C[MxN] = A[MxK] @ B[KxN], B pre-transposed ----
// MODE: 0 = f32 row-major, 1 = bf16 row-major, 2 = bf16 transposed, 3 = both bf16.
template <int HAS_BIAS, int USE_GELU, int MODE>
__global__ __launch_bounds__(256) void gemm_bf16_kernel(
    const unsigned short* __restrict__ Au, const unsigned short* __restrict__ BTu,
    const float* __restrict__ bias, float* __restrict__ Cf,
    unsigned short* __restrict__ Cbu, unsigned short* __restrict__ Cbtu,
    int M, int N, int K) {
  const bf16* A = (const bf16*)Au;
  const bf16* BT = (const bf16*)BTu;  // [N][K]
  bf16* Cb = (bf16*)Cbu;
  bf16* Cbt = (bf16*)Cbtu;
  __shared__ alignas(32) bf16 aT[128 * 32];
  __shared__ alignas(32) bf16 bT[64 * 32];  // [n][k]: fragments contiguous
  int tid = threadIdx.x, w = tid >> 5, lane = tid & 31;
  int col = lane & 15, half = lane >> 4;
  int bm = blockIdx.x, bn = blockIdx.y;
  v8f acc[4];
#pragma unroll
  for (int t = 0; t < 4; ++t)
#pragma unroll
    for (int u = 0; u < 8; ++u) acc[t][u] = 0.f;

  int arow = tid >> 1, ac0 = (tid & 1) * 16;
  int brow = tid >> 2, bc0 = (tid & 3) * 8;
  const bf16* agp = A + (size_t)(bm * 128 + arow) * K + ac0;
  const bf16* bgp = BT + (size_t)(bn * 64 + brow) * K + bc0;

  for (int k0 = 0; k0 < K; k0 += 32) {
    __syncthreads();
    {
      const uint4* as = (const uint4*)(agp + k0);
      uint4* ad = (uint4*)(aT + arow * 32 + ac0);
      ad[0] = as[0];
      ad[1] = as[1];
      *(uint4*)(bT + brow * 32 + bc0) = *(const uint4*)(bgp + k0);
      if (k0 + 32 < K) {
        __builtin_prefetch(agp + k0 + 32, 0, 1);  // global_prefetch_b8
        __builtin_prefetch(bgp + k0 + 32, 0, 1);
      }
    }
    __syncthreads();
    v16bf aF = load_a16x32(aT + (size_t)w * 16 * 32, 32);
#pragma unroll
    for (int t = 0; t < 4; ++t) {
      v16bf bF = load_b32x16_tr(bT + (size_t)t * 16 * 32, 32);
      acc[t] = wmma_bf16f32(aF, bF, acc[t]);
    }
  }
#pragma unroll
  for (int t = 0; t < 4; ++t) {
    int cn = bn * 64 + t * 16 + col;
    float bv = HAS_BIAS ? bias[cn] : 0.f;
#pragma unroll
    for (int u = 0; u < 8; ++u) {
      int rr = bm * 128 + w * 16 + u + half * 8;
      float xv = acc[t][u] + bv;
      if (USE_GELU) xv = gelu_exact(xv);
      if (MODE == 0) Cf[(size_t)rr * N + cn] = xv;
      if (MODE == 1 || MODE == 3) Cb[(size_t)rr * N + cn] = (bf16)xv;
      if (MODE == 2 || MODE == 3) Cbt[(size_t)cn * M + rr] = (bf16)xv;
    }
  }
}

// ---------------- segment-recurrent attention ----------------
// q,k row-major [tok][512]; kt,vt transposed [512][MTOT] (per-head 64-row bands).
__global__ __launch_bounds__(256) void attn_kernel(
    const unsigned short* __restrict__ qu, const unsigned short* __restrict__ ku,
    const unsigned short* __restrict__ ktu, const unsigned short* __restrict__ vtu,
    const float* __restrict__ betas, unsigned short* __restrict__ au) {
  const bf16* qg = (const bf16*)qu;
  const bf16* kg = (const bf16*)ku;
  const bf16* ktg = (const bf16*)ktu;
  const bf16* vtg = (const bf16*)vtu;
  bf16* acat = (bf16*)au;

  int b = blockIdx.x >> 3, h = blockIdx.x & 7;
  int tid = threadIdx.x, w = tid >> 5, lane = tid & 31;
  int col = lane & 15, half = lane >> 4;

  __shared__ alignas(32) bf16 k_lds[128 * DKV];   // chunk, row-major [s][d]
  __shared__ alignas(32) bf16 skt_lds[DKV * 128]; // chunk, [d][s]; elu'd in place
  __shared__ alignas(32) bf16 vt_lds[DKV * 128];  // chunk, [dv][s]
  __shared__ alignas(32) bf16 mem_bT[DKV * DKV];  // mem transposed [dv][dk] bf16
  __shared__ float mem_f[DKV * DKV];              // f32 master [dk][dv]
  __shared__ float z_f[DKV];
  __shared__ float red[8 * 16];
  __shared__ alignas(32) bf16 p_lds[8 * 16 * 32]; // per-wave P relayout tile

  for (int i = tid; i < DKV * DKV; i += 256) mem_f[i] = 0.f;
  for (int i = tid; i < DKV * DKV; i += 256) mem_bT[i] = (bf16)0.f;
  if (tid < DKV) z_f[tid] = 1.f / (float)DKV;
  __syncthreads();

#if USE_TDM
  unsigned k_off = (unsigned)(size_t)(const void*)k_lds;
  unsigned skt_off = (unsigned)(size_t)(const void*)skt_lds;
  unsigned vt_off = (unsigned)(size_t)(const void*)vt_lds;
#endif

  size_t hoff = (size_t)h * DKV;
  size_t btok = (size_t)b * SSEL;

  for (int seg = 0; seg < 8; ++seg) {
    size_t stok = btok + (size_t)seg * SEGC;

    // ================= attention phase (pre-update mem, z) =================
    for (int qi = 0; qi < 4; ++qi) {
      int qt = qi * 8 + w;
      size_t qrow = stok + (size_t)qt * 16;

      v16bf qf[2];
#pragma unroll
      for (int ks = 0; ks < 2; ++ks)
        qf[ks] = load_a16x32(qg + qrow * DD + hoff + ks * 32, DD);

      float m8[8], l8[8], alpha[8];
      v8f o[4];
#pragma unroll
      for (int u = 0; u < 8; ++u) { m8[u] = -3.0e38f; l8[u] = 0.f; }
#pragma unroll
      for (int t = 0; t < 4; ++t)
#pragma unroll
        for (int u = 0; u < 8; ++u) o[t][u] = 0.f;

      for (int kc = 0; kc < 4; ++kc) {
        size_t tok0 = stok + (size_t)kc * 128;
        __syncthreads();
#if USE_TDM
        if (w == 0) {  // one wave drives the DMA; EXEC-independent TDM op
          tdm_load_2d(k_off, kg + tok0 * DD + hoff, DKV, 128, DD, MTOT, DD);
          tdm_load_2d(vt_off, vtg + hoff * MTOT + tok0, 128, DKV, MTOT, DD, MTOT);
          __builtin_amdgcn_s_wait_tensorcnt(0);
        }
#else
        load_rm_chunk(kg, k_lds, tok0, hoff);
        load_tr_chunk(vtg, vt_lds, tok0, hoff);
#endif
        __syncthreads();

        for (int kt = 0; kt < 4; ++kt) {  // 32-key groups in the 128-key chunk
          v8f sacc[2];
#pragma unroll
          for (int sub = 0; sub < 2; ++sub) {
#pragma unroll
            for (int u = 0; u < 8; ++u) sacc[sub][u] = 0.f;
            const bf16* kb_base = k_lds + (size_t)(kt * 32 + sub * 16) * DKV;
#pragma unroll
            for (int ks = 0; ks < 2; ++ks) {
              v16bf bF = load_b32x16_tr(kb_base + ks * 32, DKV);  // B[d][key]
              sacc[sub] = wmma_bf16f32(qf[ks], bF, sacc[sub]);
            }
          }
          // ---- online softmax over 32 keys (row spans 16 lanes) ----
#pragma unroll
          for (int u = 0; u < 8; ++u) {
            float s0 = sacc[0][u] * 0.125f, s1 = sacc[1][u] * 0.125f;
            sacc[0][u] = s0; sacc[1][u] = s1;
            float mx = fmaxf(s0, s1);
#pragma unroll
            for (int msk = 8; msk >= 1; msk >>= 1)
              mx = fmaxf(mx, __shfl_xor(mx, msk, 32));
            float mn = fmaxf(m8[u], mx);
            alpha[u] = __expf(m8[u] - mn);
            m8[u] = mn;
          }
          bf16* pw = p_lds + (size_t)w * (16 * 32);
#pragma unroll
          for (int sub = 0; sub < 2; ++sub) {
#pragma unroll
            for (int u = 0; u < 8; ++u) {
              float p = __expf(sacc[sub][u] - m8[u]);
              float rs = p;
#pragma unroll
              for (int msk = 8; msk >= 1; msk >>= 1) rs += __shfl_xor(rs, msk, 32);
              l8[u] = l8[u] * (sub == 0 ? alpha[u] : 1.f) + rs;
              pw[(u + half * 8) * 32 + sub * 16 + col] = (bf16)p;
            }
          }
#pragma unroll
          for (int t = 0; t < 4; ++t)
#pragma unroll
            for (int u = 0; u < 8; ++u) o[t][u] *= alpha[u];
          // ---- O += P(16x32) @ V(32x64) : V fragments from transposed store ----
          v16bf pF = load_a16x32(pw, 32);
#pragma unroll
          for (int t = 0; t < 4; ++t) {
            v16bf vF = load_b32x16_tr(vt_lds + (size_t)(t * 16) * 128 + kt * 32, 128);
            o[t] = wmma_bf16f32(pF, vF, o[t]);
          }
        }
      }

      // ---- epilogue: att_mem = (sq@mem)/(sq@z); gate-combine; store ----
      float dpart = 0.f;
      v16bf sqF[2];
#pragma unroll
      for (int ks = 0; ks < 2; ++ks) {
#pragma unroll
        for (int u = 0; u < 16; ++u) {
          float xv = elu1f((float)qf[ks][u]);
          sqF[ks][u] = (bf16)xv;
          int kk = ks * 32 + half * 8 + ((u < 8) ? u : 8 + u);  // +16 for hi group
          dpart += xv * z_f[kk];
        }
      }
      v8f num[4];
#pragma unroll
      for (int t = 0; t < 4; ++t)
#pragma unroll
        for (int u = 0; u < 8; ++u) num[t][u] = 0.f;
#pragma unroll
      for (int ks = 0; ks < 2; ++ks)
#pragma unroll
        for (int t = 0; t < 4; ++t) {
          v16bf mF = load_b32x16_tr(mem_bT + (size_t)(t * 16) * DKV + ks * 32, DKV);
          num[t] = wmma_bf16f32(sqF[ks], mF, num[t]);
        }
      dpart += __shfl_xor(dpart, 16, 32);
      if (half == 0) red[w * 16 + col] = dpart;  // wave-private slice
      float denom[8];
#pragma unroll
      for (int u = 0; u < 8; ++u) denom[u] = red[w * 16 + u + half * 8];

#pragma unroll
      for (int t = 0; t < 4; ++t) {
        int cn = t * 16 + col;
        float beta = betas[h * DKV + cn];
        float g = 1.f / (1.f + __expf(-beta));
#pragma unroll
        for (int u = 0; u < 8; ++u) {
          float attdot = o[t][u] / l8[u];
          float attmem = num[t][u] / denom[u];
          float val = g * attmem + (1.f - g) * attdot;
          acat[(qrow + u + half * 8) * DD + hoff + cn] = (bf16)val;
        }
      }
    }

    // ======== memory update: mem += sk^T @ v ; z += sum_s sk ========
    __syncthreads();
    {
      int tile = w * 2;  // 16 (dk x dv) output tiles, 2 per wave
      v8f macc[2];
#pragma unroll
      for (int i = 0; i < 2; ++i)
#pragma unroll
        for (int u = 0; u < 8; ++u) macc[i][u] = 0.f;
      float zacc = 0.f;

      for (int kc = 0; kc < 4; ++kc) {
        size_t tok0 = stok + (size_t)kc * 128;
        __syncthreads();
#if USE_TDM
        if (w == 0) {
          tdm_load_2d(skt_off, ktg + hoff * MTOT + tok0, 128, DKV, MTOT, DD, MTOT);
          tdm_load_2d(vt_off, vtg + hoff * MTOT + tok0, 128, DKV, MTOT, DD, MTOT);
          __builtin_amdgcn_s_wait_tensorcnt(0);
        }
#else
        load_tr_chunk(ktg, skt_lds, tok0, hoff);
        load_tr_chunk(vtg, vt_lds, tok0, hoff);
#endif
        __syncthreads();
        // sk = elu(k)+1, computed once per chunk, in place, contiguous.
        for (int i = tid; i < (DKV * 128) / 2; i += 256) {
          unsigned* p = (unsigned*)skt_lds + i;
          unsigned pv = *p;
          bf16 b0 = ((const bf16*)&pv)[0], b1 = ((const bf16*)&pv)[1];
          bf16 r2[2] = {(bf16)elu1f((float)b0), (bf16)elu1f((float)b1)};
          *p = *(const unsigned*)r2;
        }
        __syncthreads();
#pragma unroll
        for (int i = 0; i < 2; ++i) {
          int dkt = (tile + i) >> 2, dvt = (tile + i) & 3;
          for (int s0 = 0; s0 < 128; s0 += 32) {
            v16bf aF = load_a16x32(skt_lds + (size_t)(dkt * 16) * 128 + s0, 128);
            v16bf vF = load_b32x16_tr(vt_lds + (size_t)(dvt * 16) * 128 + s0, 128);
            macc[i] = wmma_bf16f32(aF, vF, macc[i]);
          }
        }
        if (tid < DKV) {
          const bf16* sr = skt_lds + (size_t)tid * 128;
          for (int s = 0; s < 128; ++s) zacc += (float)sr[s];
        }
        __syncthreads();  // protect skt/vt before next chunk DMA
      }
      __syncthreads();
#pragma unroll
      for (int i = 0; i < 2; ++i) {
        int dkt = (tile + i) >> 2, dvt = (tile + i) & 3;
#pragma unroll
        for (int u = 0; u < 8; ++u) {
          int rr = dkt * 16 + u + half * 8;
          int cc = dvt * 16 + col;
          mem_f[rr * DKV + cc] += macc[i][u];  // unique writer per element
        }
      }
      if (tid < DKV) z_f[tid] += zacc;
      __syncthreads();
      // regenerate transposed bf16 mirror for the next segment's WMMA reads
      for (int i = tid; i < DKV * DKV; i += 256) {
        int dk = i & 63, dv = i >> 6;
        mem_bT[i] = (bf16)mem_f[dk * DKV + dv];  // mem_bT[dv][dk]
      }
      __syncthreads();
    }
  }
}

// x_out = (s < SSEL ? x : 0); mask = 0 (scatter sets 1s afterwards).
__global__ __launch_bounds__(128) void copy_pad_kernel(const float* __restrict__ x,
                                                       float* __restrict__ x_out,
                                                       float* __restrict__ mask) {
  size_t row = blockIdx.x;
  int s = (int)(row % SS);
  const float4* src = (const float4*)(x + row * DD);
  float4* dst = (float4*)(x_out + row * DD);
  float4 z4 = {0.f, 0.f, 0.f, 0.f};
  dst[threadIdx.x] = (s < SSEL) ? src[threadIdx.x] : z4;
  if (threadIdx.x == 0) mask[row] = 0.f;
}

__global__ __launch_bounds__(128) void scatter_kernel(const float* __restrict__ h,
                                                      const int* __restrict__ flat,
                                                      float* __restrict__ x_out,
                                                      float* __restrict__ mask) {
  int r = blockIdx.x;
  int b = r / SSEL;
  int dst = flat[r];  // unique per batch -> no atomics
  if (threadIdx.x == 0) mask[(size_t)b * SS + dst] = 1.f;
  if (dst < SSEL) {
    const float4* hp = (const float4*)(h + (size_t)r * DD);
    float4* xp = (float4*)(x_out + ((size_t)b * SS + dst) * DD);
    float4 a = xp[threadIdx.x];
    float4 hv = hp[threadIdx.x];
    a.x += hv.x; a.y += hv.y; a.z += hv.z; a.w += hv.w;
    xp[threadIdx.x] = a;
  }
}

// ---------------- host orchestration ----------------
extern "C" void kernel_launch(void* const* d_in, const int* in_sizes, int n_in,
                              void* d_out, int out_size, void* d_ws, size_t ws_size,
                              hipStream_t stream) {
  (void)in_sizes; (void)n_in; (void)out_size; (void)ws_size;
  const float* x      = (const float*)d_in[0];
  const float* w_samp = (const float*)d_in[1];
  const float* b_samp = (const float*)d_in[2];
  const float* w_q    = (const float*)d_in[3];
  const float* w_k    = (const float*)d_in[4];
  const float* w_v    = (const float*)d_in[5];
  const float* w_o    = (const float*)d_in[6];
  const float* betas  = (const float*)d_in[7];
  const float* w1     = (const float*)d_in[8];
  const float* b1     = (const float*)d_in[9];
  const float* w2     = (const float*)d_in[10];
  const float* b2     = (const float*)d_in[11];

  float* x_out  = (float*)d_out;
  float* mask   = x_out + (size_t)BB * SS * DD;
  float* scores = mask + (size_t)BB * SS;

  char* wsp = (char*)d_ws;
  auto carve = [&](size_t bytes) -> char* {
    char* p = wsp;
    wsp += (bytes + 255) & ~(size_t)255;
    return p;
  };
  unsigned short* wqT = (unsigned short*)carve((size_t)DD * DD * 2);
  unsigned short* wkT = (unsigned short*)carve((size_t)DD * DD * 2);
  unsigned short* wvT = (unsigned short*)carve((size_t)DD * DD * 2);
  unsigned short* woT = (unsigned short*)carve((size_t)DD * DD * 2);
  unsigned short* w1T = (unsigned short*)carve((size_t)DD * DHID * 2);
  unsigned short* w2T = (unsigned short*)carve((size_t)DHID * DD * 2);
  int* flat            = (int*)carve((size_t)MTOT * 4);
  unsigned short* xsel = (unsigned short*)carve((size_t)MTOT * DD * 2);
  unsigned short* qb   = (unsigned short*)carve((size_t)MTOT * DD * 2);
  unsigned short* kb   = (unsigned short*)carve((size_t)MTOT * DD * 2);  // row-major
  unsigned short* ktb  = (unsigned short*)carve((size_t)MTOT * DD * 2);  // [D][M]
  unsigned short* vtb  = (unsigned short*)carve((size_t)MTOT * DD * 2);  // [D][M]
  unsigned short* acat = (unsigned short*)carve((size_t)MTOT * DD * 2);
  unsigned short* h0   = (unsigned short*)carve((size_t)MTOT * DD * 2);
  unsigned short* h1   = (unsigned short*)carve((size_t)MTOT * DHID * 2);
  float* hf            = (float*)carve((size_t)MTOT * DD * 4);

  // 1) weights -> bf16, pre-transposed to [N][K]
  convert_transpose_kernel<<<(DD * DD + 255) / 256, 256, 0, stream>>>(w_q, wqT, DD, DD);
  convert_transpose_kernel<<<(DD * DD + 255) / 256, 256, 0, stream>>>(w_k, wkT, DD, DD);
  convert_transpose_kernel<<<(DD * DD + 255) / 256, 256, 0, stream>>>(w_v, wvT, DD, DD);
  convert_transpose_kernel<<<(DD * DD + 255) / 256, 256, 0, stream>>>(w_o, woT, DD, DD);
  convert_transpose_kernel<<<(DD * DHID + 255) / 256, 256, 0, stream>>>(w1, w1T, DD, DHID);
  convert_transpose_kernel<<<(DHID * DD + 255) / 256, 256, 0, stream>>>(w2, w2T, DHID, DD);

  // 2) router scores (written straight into the scores output slot)
  router_kernel<<<(BB * SS) / 8, 256, 0, stream>>>(x, w_samp, b_samp, scores);

  // 3) per-segment top-k routing, 4) gather
  topk_kernel<<<BB * NSEGF, 1024, 0, stream>>>(scores, flat);
  gather_kernel<<<MTOT, 128, 0, stream>>>(x, flat, xsel);

  // 5) QKV projections: Q row-major; K both layouts; V transposed only
  dim3 g512(MTOT / 128, DD / 64);
  gemm_bf16_kernel<0, 0, 1><<<g512, 256, 0, stream>>>(xsel, wqT, nullptr, nullptr, qb, nullptr, MTOT, DD, DD);
  gemm_bf16_kernel<0, 0, 3><<<g512, 256, 0, stream>>>(xsel, wkT, nullptr, nullptr, kb, ktb, MTOT, DD, DD);
  gemm_bf16_kernel<0, 0, 2><<<g512, 256, 0, stream>>>(xsel, wvT, nullptr, nullptr, nullptr, vtb, MTOT, DD, DD);

  // 6) compressive-memory attention (segment recurrence per (b, head))
  attn_kernel<<<BB * NH, 256, 0, stream>>>(qb, kb, ktb, vtb, betas, acat);

  // 7) output projection, 8) MLP up (+bias, GELU), 9) MLP down (+bias, f32)
  gemm_bf16_kernel<0, 0, 1><<<g512, 256, 0, stream>>>(acat, woT, nullptr, nullptr, h0, nullptr, MTOT, DD, DD);
  dim3 g2048(MTOT / 128, DHID / 64);
  gemm_bf16_kernel<1, 1, 1><<<g2048, 256, 0, stream>>>(h0, w1T, b1, nullptr, h1, nullptr, MTOT, DHID, DD);
  gemm_bf16_kernel<1, 0, 0><<<g512, 256, 0, stream>>>(h1, w2T, b2, hf, nullptr, nullptr, MTOT, DD, DHID);

  // 10) copy x with pad-zeroing + clear mask, 11) scatter-add + set mask
  copy_pad_kernel<<<BB * SS, 128, 0, stream>>>(x, x_out, mask);
  scatter_kernel<<<MTOT, 128, 0, stream>>>(hf, flat, x_out, mask);
}
